// RowChooserLayer_71648644432065
// MI455X (gfx1250) — matrix operand order repaired and checked by reading, not compiled, not run on hardware
//
#include <hip/hip_runtime.h>

typedef __attribute__((ext_vector_type(16))) _Float16 v16h;
typedef __attribute__((ext_vector_type(8)))  float    v8f;
typedef __attribute__((ext_vector_type(4)))  float    f4;

#define BB 32
#define NN 4096
#define DD 256
#define FF 256

// ---------------------------------------------------------------------------
// Kernel 1: x = relu(A @ W + bias); mask = x > threshold
// A: [32,256] row-major, W: [256,4096] row-major. One wave per 16x16 tile of
// the [32,4096] output; K=256 consumed as 8 x v_wmma_f32_16x16x32_f16.
// ---------------------------------------------------------------------------
__global__ __launch_bounds__(128) void gemm_mask_kernel(
    const float* __restrict__ A,
    const float* __restrict__ W,
    const float* __restrict__ bias,
    const float* __restrict__ thr,
    int* __restrict__ maskws)
{
    const int wave = (blockIdx.x * blockDim.x + threadIdx.x) >> 5;  // 0..511
    const int lane = threadIdx.x & 31;
    const int l    = lane & 15;     // sub-lane within half
    const int lh   = lane >> 4;     // lane half (K-stripe selector)

    const int m0 = (wave >> 8) * 16;   // 0 or 16
    const int n0 = (wave & 255) * 16;  // 0..4080

    const int m = m0 + l;   // A row for this lane (both halves cover M=0..15)
    const int n = n0 + l;   // B/C column for this lane

    v8f c = {};
    for (int kk = 0; kk < FF; kk += 32) {
        v16h a, bm;
        // A 16x32 f16 layout: VGPR v<4 -> K block 0..15, v>=4 -> 16..31;
        // within a 16-block, lane-half picks the 8-wide sub-stripe.
        #pragma unroll
        for (int v = 0; v < 8; ++v) {
            const int kA = kk + ((v < 4) ? 0 : 16) + lh * 8 + ((v & 3) * 2);
            const float2 av = *(const float2*)(A + m * FF + kA);
            a[2 * v]     = (_Float16)av.x;
            a[2 * v + 1] = (_Float16)av.y;
            // B 32x16 f16 layout: lane = column, same K order as A:
            // v<4 -> K 0..15 block, v>=4 -> 16..31, lane-half picks 8-wide sub-stripe.
            const int kB = kk + ((v < 4) ? 0 : 16) + lh * 8 + ((v & 3) * 2);
            bm[2 * v]     = (_Float16)W[kB * NN + n];
            bm[2 * v + 1] = (_Float16)W[(kB + 1) * NN + n];
        }
        c = __builtin_amdgcn_wmma_f32_16x16x32_f16(
                false, a, false, bm, (short)0, c, false, false);
    }

    const float bv = bias[n];
    const float th = thr[0];
    // C/D layout: VGPR v -> M = v (lanes 0-15) or M = v+8 (lanes 16-31)
    #pragma unroll
    for (int v = 0; v < 8; ++v) {
        const int row = m0 + lh * 8 + v;          // == batch index (M=B=32)
        float x = c[v] + bv;
        x = fmaxf(x, 0.0f);
        maskws[row * NN + n] = (x > th) ? 1 : 0;
    }
}

// ---------------------------------------------------------------------------
// Kernel 2: per-batch stable scan. For each row n emit an encoded destination:
//   selected:   enc = (#selected in [0,n))                  (>= 0)
//   unselected: enc = -1 - (count + #unselected in [0,n))   (< 0, zero-fill)
// Every output row is therefore produced exactly once by kernel 3.
// ---------------------------------------------------------------------------
__global__ __launch_bounds__(1024) void scan_kernel(
    const int* __restrict__ maskws,
    int* __restrict__ dst)
{
    __shared__ int sh[1024];
    const int b   = blockIdx.x;
    const int tid = threadIdx.x;
    const int* mrow = maskws + b * NN;

    int m4[4];
    int s = 0;
    #pragma unroll
    for (int i = 0; i < 4; ++i) { m4[i] = mrow[tid * 4 + i]; s += m4[i]; }

    sh[tid] = s;
    __syncthreads();
    for (int off = 1; off < 1024; off <<= 1) {
        const int v = (tid >= off) ? sh[tid - off] : 0;
        __syncthreads();
        sh[tid] += v;
        __syncthreads();
    }
    const int total = sh[1023];        // count of selected rows in batch b
    int run = sh[tid] - s;             // exclusive prefix of selected

    int* drow = dst + b * NN;
    #pragma unroll
    for (int i = 0; i < 4; ++i) {
        const int n = tid * 4 + i;
        drow[n] = m4[i] ? run : (-1 - (total + (n - run)));
        run += m4[i];
    }
}

// ---------------------------------------------------------------------------
// Kernel 3: streaming gather. 64 lanes move one 1KB row as float4 (B128),
// nontemporal on both sides (single-touch data, don't churn L2).
// ---------------------------------------------------------------------------
__global__ __launch_bounds__(256) void gather_kernel(
    const float* __restrict__ raw,
    const int* __restrict__ dst,
    float* __restrict__ out)
{
    const int grp = threadIdx.x >> 6;               // 4 rows per block
    const int t   = threadIdx.x & 63;               // 64 lanes * float4 = 256 f
    const int gr  = blockIdx.x * 4 + grp;           // global row 0..131071
    const int b   = gr >> 12;
    const int enc = dst[gr];

    if (enc >= 0) {
        const f4* src = (const f4*)(raw + (size_t)gr * DD);
        f4* o = (f4*)(out + ((size_t)(b << 12) + (size_t)enc) * DD);
        const f4 v = __builtin_nontemporal_load(src + t);
        __builtin_nontemporal_store(v, o + t);
    } else {
        const int row = -1 - enc;
        f4* o = (f4*)(out + ((size_t)(b << 12) + (size_t)row) * DD);
        const f4 z = {0.0f, 0.0f, 0.0f, 0.0f};
        __builtin_nontemporal_store(z, o + t);
    }
}

extern "C" void kernel_launch(void* const* d_in, const int* in_sizes, int n_in,
                              void* d_out, int out_size, void* d_ws, size_t ws_size,
                              hipStream_t stream) {
    const float* raw  = (const float*)d_in[0];  // [32,4096,256]
    const float* ipa  = (const float*)d_in[1];  // [32,256]
    const float* W    = (const float*)d_in[2];  // [256,4096]
    const float* bias = (const float*)d_in[3];  // [4096]
    const float* thr  = (const float*)d_in[4];  // [1]
    float* out = (float*)d_out;                 // [32,4096,256]

    int* maskws = (int*)d_ws;                   // 32*4096 ints
    int* dstws  = maskws + BB * NN;             // 32*4096 ints (1 MB total)

    // 512 waves = 2 M-tiles x 256 N-tiles
    gemm_mask_kernel<<<128, 128, 0, stream>>>(ipa, W, bias, thr, maskws);
    // one block per batch
    scan_kernel<<<BB, 1024, 0, stream>>>(maskws, dstws);
    // 4 rows per 256-thread block
    gather_kernel<<<(BB * NN) / 4, 256, 0, stream>>>(raw, dstws, out);
}
